// DCABlock_28260884808333
// MI455X (gfx1250) — matrix operand order repaired and checked by reading, not compiled, not run on hardware
//
#include <hip/hip_runtime.h>
#include <hip/hip_bf16.h>
#include <stdint.h>

// ---------------------------------------------------------------------------
// DCA block for MI455X (gfx1250, wave32).
//  * All GEMM operands pre-converted to bf16 and stored K-major
//    (A as [M][K], B as [N][K]) -> LDS tiles are contiguous byte copies.
//  * GEMM global->LDS staging uses GLOBAL_LOAD_ASYNC_TO_LDS_B128
//    (ASYNCcnt, double-buffered; copy of tile k+1 overlaps WMMA on tile k).
//  * Math: v_wmma_f32_16x16x32_bf16, fp32 accumulation.
//  * Dead reference branch (w_theta / w_eca_k) skipped.
// ---------------------------------------------------------------------------

typedef __attribute__((ext_vector_type(16))) __bf16 v16bf;
typedef __attribute__((ext_vector_type(8)))  float  v8f;

struct Bits256 { uint4 lo, hi; };   // 32 bytes == v16bf

__device__ __forceinline__ unsigned short f2bf(float f) {
    unsigned u = __builtin_bit_cast(unsigned, f);
    u += 0x7fffu + ((u >> 16) & 1u);          // round-to-nearest-even
    return (unsigned short)(u >> 16);
}

// ---------------------------------------------------------------------------
// WMMA GEMM, operands bf16 K-major: C(MxN) = A[M][K] . B[N][K]^T_logical
//   block = 128 threads = 4 waves; block tile 128x128; wave tile 64x64; BK=32
//   EPI: 0: C.f32[m,n] = acc
//        1: C.bf16[m,n] = acc + E[m,n]   (fused addT = A + Bm, bf16 out)
//        2: C.f32[m,n]  = acc + E[m,n]   (fused +x residual)
// ---------------------------------------------------------------------------
template <int EPI>
__global__ __launch_bounds__(128)
void gemm128_wmma_async(const unsigned short* __restrict__ A,
                        const unsigned short* __restrict__ B,
                        void* __restrict__ Cv, const float* __restrict__ E,
                        int M, int N, int K, int ldc,
                        long long sA, long long sB, long long sC, long long sE)
{
    const int b = blockIdx.z;
    A += (size_t)b * sA;
    B += (size_t)b * sB;
    if (E) E += (size_t)b * sE;

    const int m0 = blockIdx.y * 128;
    const int n0 = blockIdx.x * 128;

    // [buf][A|B][128 rows x 32 bf16]  (8 KB per tile, 32 KB total)
    __shared__ alignas(16) unsigned short sT[2][2][128 * 32];

    const int tid  = threadIdx.x;
    const int lane = tid & 31;
    const int wv   = tid >> 5;
    const int wm   = (wv >> 1) * 64;
    const int wn   = (wv & 1) * 64;

    // ---- per-thread async-copy addressing: 4 chunks of 16B for A and B ----
    const int rr = tid >> 2;          // row group 0..31
    const int q  = tid & 3;           // 16B chunk within 64B row
    uint64_t gA[4], gB[4];
    unsigned lOff[4];
#pragma unroll
    for (int i = 0; i < 4; ++i) {
        gA[i]   = (uint64_t)(uintptr_t)A + ((uint64_t)(m0 + rr + 32 * i) * K) * 2 + q * 16;
        gB[i]   = (uint64_t)(uintptr_t)B + ((uint64_t)(n0 + rr + 32 * i) * K) * 2 + q * 16;
        lOff[i] = (unsigned)((rr + 32 * i) * 64 + q * 16);
    }
    const unsigned ldsBase = (unsigned)(uintptr_t)&sT[0][0][0]; // low 32 bits = LDS offset

    auto issue = [&](int buf, int kByte) {
        const unsigned bA = ldsBase + (unsigned)buf * 16384u;
        const unsigned bB = bA + 8192u;
#pragma unroll
        for (int i = 0; i < 4; ++i) {
            unsigned la = bA + lOff[i];
            uint64_t ga = gA[i] + (unsigned)kByte;
            asm volatile("global_load_async_to_lds_b128 %0, %1, off"
                         :: "v"(la), "v"(ga) : "memory");
            unsigned lb = bB + lOff[i];
            uint64_t gb = gB[i] + (unsigned)kByte;
            asm volatile("global_load_async_to_lds_b128 %0, %1, off"
                         :: "v"(lb), "v"(gb) : "memory");
        }
    };

    v8f acc[4][4];
#pragma unroll
    for (int i = 0; i < 4; ++i)
#pragma unroll
        for (int j = 0; j < 4; ++j)
#pragma unroll
            for (int e = 0; e < 8; ++e) acc[i][j][e] = 0.0f;

    const int mA = lane & 15;           // fragment row within 16
    const int kh = lane >> 4;           // K-half selector (ISA 16-bit layout)

    const int nk = K >> 5;
    issue(0, 0);                        // prefetch first tile pair

    for (int kt = 0; kt < nk; ++kt) {
        const int cur = kt & 1;
        asm volatile("s_wait_asynccnt 0x0" ::: "memory"); // own copies to cur done
        __syncthreads();                                  // all waves' copies done,
                                                          // all readers of cur^1 done
        if (kt + 1 < nk) issue(cur ^ 1, (kt + 1) << 6);   // overlap next copy (bytes = k0*2)

        const unsigned short* tA = &sT[cur][0][0];
        const unsigned short* tB = &sT[cur][1][0];

        v16bf aF[4], bF[4];
#pragma unroll
        for (int i = 0; i < 4; ++i) {
            int base = (wm + i * 16 + mA) * 32 + kh * 8;
            Bits256 bits;
            bits.lo = *(const uint4*)&tA[base];           // K = kh*8 + 0..7
            bits.hi = *(const uint4*)&tA[base + 16];      // K = 16 + kh*8 + 0..7
            aF[i] = __builtin_bit_cast(v16bf, bits);
        }
#pragma unroll
        for (int j = 0; j < 4; ++j) {
            int base = (wn + j * 16 + (lane & 15)) * 32 + kh * 8;
            Bits256 bits;
            bits.lo = *(const uint4*)&tB[base];           // K = kh*8 + 0..7
            bits.hi = *(const uint4*)&tB[base + 16];      // K = 16 + kh*8 + 0..7
            bF[j] = __builtin_bit_cast(v16bf, bits);
        }

#pragma unroll
        for (int i = 0; i < 4; ++i)
#pragma unroll
            for (int j = 0; j < 4; ++j)
                acc[i][j] = __builtin_amdgcn_wmma_f32_16x16x32_bf16(
                    false, aF[i], false, bF[j], (short)0, acc[i][j],
                    false, false);
    }

    // ---- epilogue: D layout — lane&15 = N col, row = vgpr + (lane>>4)*8 ----
    const int cn = lane & 15;
    const int rb = (lane >> 4) * 8;
#pragma unroll
    for (int i = 0; i < 4; ++i)
#pragma unroll
        for (int j = 0; j < 4; ++j)
#pragma unroll
            for (int r = 0; r < 8; ++r) {
                int m = m0 + wm + i * 16 + rb + r;
                int n = n0 + wn + j * 16 + cn;
                float v = acc[i][j][r];
                if (EPI == 0) {
                    ((float*)Cv + (size_t)b * sC)[(size_t)m * ldc + n] = v;
                } else if (EPI == 1) {
                    ((unsigned short*)Cv + (size_t)b * sC)[(size_t)m * ldc + n] =
                        f2bf(v + E[(size_t)m * ldc + n]);
                } else {
                    ((float*)Cv + (size_t)b * sC)[(size_t)m * ldc + n] =
                        v + E[(size_t)m * ldc + n];
                }
            }
}

// ---------------------------------------------------------------------------
// Column softmax over axis=1 (row index); fp32 in, bf16 out.
// TRANSOUT=0: dst[r*cols+col]; TRANSOUT=1: dst[col*rows+r] (K-major for GEMM).
// ---------------------------------------------------------------------------
template <int TRANSOUT>
__global__ void col_softmax_bf16(const float* __restrict__ src,
                                 unsigned short* __restrict__ dst,
                                 int rows, int cols, long long sIn, long long sOut)
{
    const int b = blockIdx.y;
    src += (size_t)b * sIn;
    dst += (size_t)b * sOut;
    const int col = blockIdx.x * blockDim.x + threadIdx.x;
    if (col >= cols) return;

    float mx = -3.4e38f;
    for (int r = 0; r < rows; ++r) mx = fmaxf(mx, src[(size_t)r * cols + col]);
    float s = 0.0f;
    for (int r = 0; r < rows; ++r) s += __expf(src[(size_t)r * cols + col] - mx);
    const float inv = 1.0f / s;
    for (int r = 0; r < rows; ++r) {
        unsigned short v = f2bf(__expf(src[(size_t)r * cols + col] - mx) * inv);
        if (TRANSOUT) dst[(size_t)col * rows + r] = v;
        else          dst[(size_t)r * cols + col] = v;
    }
}

// Row mean: one block per row of `cols` contiguous floats.
__global__ __launch_bounds__(256)
void row_mean(const float* __restrict__ src, float* __restrict__ means, int cols)
{
    const float* p = src + (size_t)blockIdx.x * cols;
    float s = 0.0f;
    for (int i = threadIdx.x; i < cols; i += 256) s += p[i];
    __shared__ float red[256];
    red[threadIdx.x] = s;
    __syncthreads();
#pragma unroll
    for (int off = 128; off > 0; off >>= 1) {
        if (threadIdx.x < off) red[threadIdx.x] += red[threadIdx.x + off];
        __syncthreads();
    }
    if (threadIdx.x == 0) means[blockIdx.x] = red[0] / (float)cols;
}

// ECA gate: conv5 (cross-correlation, zero pad) over channel means + sigmoid.
__global__ void eca_scale(const float* __restrict__ means,
                          const float* __restrict__ w,
                          float* __restrict__ scales, int ic, int k)
{
    const int b  = blockIdx.y;
    const int ch = blockIdx.x * blockDim.x + threadIdx.x;
    if (ch >= ic) return;
    const int pad = (k - 1) / 2;
    float a = 0.0f;
    for (int t = 0; t < k; ++t) {
        int idx = ch - pad + t;
        if (idx >= 0 && idx < ic) a += w[t] * means[(size_t)b * ic + idx];
    }
    scales[(size_t)b * ic + ch] = 1.0f + 1.0f / (1.0f + __expf(-a));
}

// Flat fp32 -> bf16 convert; SCALED multiplies by scales[i >> 10] (row of 1024).
template <int SCALED>
__global__ void conv_flat(const float* __restrict__ in,
                          const float* __restrict__ scales,
                          unsigned short* __restrict__ out, int total)
{
    int i = blockIdx.x * blockDim.x + threadIdx.x;
    if (i >= total) return;
    float v = in[i];
    if (SCALED) v *= scales[i >> 10];
    out[i] = f2bf(v);
}

// LDS-tiled transpose + convert: in fp32 (R x Cc) per batch -> out bf16 (Cc x R).
// SCALED multiplies by scales[b*R + input_row]. block (32, 8).
template <int SCALED>
__global__ void transpose_conv(const float* __restrict__ in,
                               unsigned short* __restrict__ out,
                               const float* __restrict__ scales,
                               int R, int Cc, long long sIn, long long sOut)
{
    __shared__ float t[32][33];
    const int b = blockIdx.z;
    in  += (size_t)b * sIn;
    out += (size_t)b * sOut;
    const int r0 = blockIdx.y * 32, c0 = blockIdx.x * 32;
    const int tx = threadIdx.x, ty = threadIdx.y;
#pragma unroll
    for (int ii = 0; ii < 4; ++ii) {
        int r = ty + ii * 8;
        float v = in[(size_t)(r0 + r) * Cc + c0 + tx];
        if (SCALED) v *= scales[(size_t)b * R + r0 + r];
        t[r][tx] = v;
    }
    __syncthreads();
#pragma unroll
    for (int ii = 0; ii < 4; ++ii) {
        int r = ty + ii * 8;
        out[(size_t)(c0 + r) * R + r0 + tx] = f2bf(t[tx][r]);
    }
}

// ---------------------------------------------------------------------------
extern "C" void kernel_launch(void* const* d_in, const int* in_sizes, int n_in,
                              void* d_out, int out_size, void* d_ws, size_t ws_size,
                              hipStream_t stream)
{
    (void)n_in; (void)out_size; (void)ws_size;
    const float* x      = (const float*)d_in[0];  // (16, 2048, 32, 32)
    const float* w_phi  = (const float*)d_in[1];  // (1024, 2048)
    const float* w_ecaq = (const float*)d_in[2];  // (5,)
    // d_in[3] (w_theta), d_in[4] (w_eca_k): dead in the reference -> skipped
    const float* w_mask = (const float*)d_in[5];  // (2048, 1024)
    float* out = (float*)d_out;

    const int B = 16, C = 2048, IC = 1024, N = 1024;
    const long long M1 = (long long)IC * N;        // 1M elems
    const long long M2 = (long long)C * N;         // 2M elems (x / out / xT)

    // ---- workspace layout (~233 MB) ----
    char* ws = (char*)d_ws;
    float* means  = (float*)ws;                               // B*IC
    float* scales = means + B * IC;                           // B*IC
    float*          f32buf  = (float*)(ws + (1u << 20));      // Qpre->scores->A  64MB
    unsigned short* xT_bf   = (unsigned short*)((char*)f32buf + (size_t)B * M1 * 4);  // 64MB
    unsigned short* wphi_bf = (unsigned short*)((char*)xT_bf  + (size_t)B * M2 * 2);  // 4MB
    unsigned short* wmsk_bf = (unsigned short*)((char*)wphi_bf + (size_t)M2 * 2);     // 4MB
    unsigned short* qf_bf   = (unsigned short*)((char*)wmsk_bf + (size_t)M2 * 2);     // 32MB (later addT)
    unsigned short* qfT_bf  = (unsigned short*)((char*)qf_bf  + (size_t)B * M1 * 2);  // 32MB
    unsigned short* sm_bf   = (unsigned short*)((char*)qfT_bf + (size_t)B * M1 * 2);  // 32MB (later sm2T)
    unsigned short* addT_bf = qf_bf;                          // reuse (Qf dead after GEMM3)
    unsigned short* sm2T_bf = sm_bf;                          // reuse (sm dead after GEMM3)

    dim3 blk(128);
    dim3 g1k(N / 128, IC / 128, B);     // 1024x1024 per batch
    dim3 gMk(N / 128, C / 128, B);      // 2048x1024 per batch

    // ---- 0) operand conversion: bf16, K-major ----
    conv_flat<0><<<dim3((int)(M2 / 256)), dim3(256), 0, stream>>>(w_phi, nullptr, wphi_bf, (int)M2);
    conv_flat<0><<<dim3((int)(M2 / 256)), dim3(256), 0, stream>>>(w_mask, nullptr, wmsk_bf, (int)M2);
    // xT[b][n][c] = x[b][c][n]
    transpose_conv<0><<<dim3(N / 32, C / 32, B), dim3(32, 8), 0, stream>>>(
        x, xT_bf, nullptr, C, N, M2, M2);

    // ---- 1) Qpre = w_phi @ x_b   (M=1024,N=1024,K=2048) ----
    gemm128_wmma_async<0><<<g1k, blk, 0, stream>>>(
        wphi_bf, xT_bf, f32buf, nullptr, IC, N, C, N, 0, M2, M1, 0);

    // ---- 2) ECA gate ----
    row_mean<<<dim3(B * IC), dim3(256), 0, stream>>>(f32buf, means, N);
    eca_scale<<<dim3(IC / 256, B), dim3(256), 0, stream>>>(means, w_ecaq, scales, IC, in_sizes[2]);
    // Qf bf16 in both orientations (scaled)
    conv_flat<1><<<dim3((int)(B * M1 / 256)), dim3(256), 0, stream>>>(
        f32buf, scales, qf_bf, (int)(B * M1));
    transpose_conv<1><<<dim3(N / 32, IC / 32, B), dim3(32, 8), 0, stream>>>(
        f32buf, qfT_bf, scales, IC, N, M1, M1);

    // ---- 3) scores = Qf^T @ Qf  (A=B=QfT, K=IC) -> f32buf ----
    gemm128_wmma_async<0><<<g1k, blk, 0, stream>>>(
        qfT_bf, qfT_bf, f32buf, nullptr, N, N, IC, N, M1, M1, M1, 0);

    // ---- 4) sm = softmax(scores, axis=1) -> bf16 row-major ----
    col_softmax_bf16<0><<<dim3(N / 256, B), dim3(256), 0, stream>>>(
        f32buf, sm_bf, N, N, M1, M1);

    // ---- 5) A = sm @ Qf^T  (A=sm[n][m], B=Qf[d][m], K=N) -> f32buf ----
    gemm128_wmma_async<0><<<g1k, blk, 0, stream>>>(
        sm_bf, qf_bf, f32buf, nullptr, N, IC, N, IC, M1, M1, M1, 0);

    // ---- 6) sm2 = softmax(A, axis=1) -> bf16 transposed [d][m] ----
    col_softmax_bf16<1><<<dim3(IC / 256, B), dim3(256), 0, stream>>>(
        f32buf, sm2T_bf, N, IC, M1, M1);

    // ---- 7) Bm = Qf^T @ sm2 ; fused: addT[n][d] = Bm[n,d] + A[n,d] (bf16) ----
    gemm128_wmma_async<1><<<g1k, blk, 0, stream>>>(
        qfT_bf, sm2T_bf, addT_bf, f32buf, N, IC, N, IC, M1, M1, M1, M1);

    // ---- 8) out = w_mask @ add + x   (M=2048,N=1024,K=1024) ----
    gemm128_wmma_async<2><<<gMk, blk, 0, stream>>>(
        wmsk_bf, addT_bf, out, x, C, N, IC, N, 0, M1, M2, M2);
}
